// InfoNCELoss_72765335929776
// MI455X (gfx1250) — matrix operand order repaired and checked by reading, not compiled, not run on hardware
//
#include <hip/hip_runtime.h>
#include <hip/hip_bf16.h>

// ---------------------------------------------------------------------------
// InfoNCE loss (B=4096, D=4096, fp32 in, scalar fp32 out) for gfx1250.
// Compute-bound GEMM (137 GFLOP) -> v_wmma_f32_16x16x32_f16 with fused
// online softmax so the 64MiB logits matrix never touches HBM.
// 32-row M-tiles + A tile staged in LDS (CDNA5 320KB) + 2Mx2N register
// blocking: 1 global_load_b128 + 1 ds_load_b128 per WMMA, and L2 B-traffic
// halved vs 16-row tiles.
// ---------------------------------------------------------------------------

typedef __attribute__((ext_vector_type(16))) _Float16 v16h;
typedef __attribute__((ext_vector_type(8)))  _Float16 v8h;
typedef __attribute__((ext_vector_type(4)))  _Float16 v4h;
typedef __attribute__((ext_vector_type(8)))  float    v8f;

#define TEMP_INV 14.285714285714286f   // 1 / 0.07

static __device__ __forceinline__ v16h cat16(v8h lo, v8h hi) {
  return __builtin_shufflevector(lo, hi, 0, 1, 2, 3, 4, 5, 6, 7,
                                         8, 9, 10, 11, 12, 13, 14, 15);
}

// ---------------------------------------------------------------------------
// Kernel 1: L2-normalize each row (fp32 -> f16). grid = (B, 2), block = 256.
// ---------------------------------------------------------------------------
__global__ __launch_bounds__(256) void nrm_f32_to_f16(
    const float* __restrict__ gSrc, const float* __restrict__ rSrc,
    _Float16* __restrict__ gDst, _Float16* __restrict__ rDst, int D)
{
  const float*  src = (blockIdx.y == 0) ? gSrc : rSrc;
  _Float16*     dst = (blockIdx.y == 0) ? gDst : rDst;
  const size_t  base = (size_t)blockIdx.x * (size_t)D;
  const float4* p4 = (const float4*)(src + base);
  v4h*          q4 = (v4h*)(dst + base);
  const int t = threadIdx.x;

  float4 v[4];
  float ss = 0.f;
#pragma unroll
  for (int i = 0; i < 4; ++i) {
    v[i] = p4[t + 256 * i];
    ss += v[i].x * v[i].x + v[i].y * v[i].y + v[i].z * v[i].z + v[i].w * v[i].w;
  }
#pragma unroll
  for (int off = 16; off; off >>= 1) ss += __shfl_xor(ss, off, 32);

  __shared__ float red[8];
  if ((t & 31) == 0) red[t >> 5] = ss;
  __syncthreads();
  float tot = 0.f;
#pragma unroll
  for (int w = 0; w < 8; ++w) tot += red[w];

  const float inv = 1.0f / fmaxf(sqrtf(tot), 1e-12f);  // matches F.normalize
#pragma unroll
  for (int i = 0; i < 4; ++i) {
    v4h h;
    h[0] = (_Float16)(v[i].x * inv);
    h[1] = (_Float16)(v[i].y * inv);
    h[2] = (_Float16)(v[i].z * inv);
    h[3] = (_Float16)(v[i].w * inv);
    q4[t + 256 * i] = h;
  }
}

// ---------------------------------------------------------------------------
// Kernel 2: fused GEMM (g @ r^T / T) + row-wise online softmax + diag pick.
// grid = B/32 blocks (32 rows each), block = 256 threads = 8 waves.
// A tile (32x4096 f16) lives in LDS (row stride padded to 4104 halves: 16B
// aligned, banks step by 4 per row -> <=2-way conflict on ds_load_b128).
// Wave w owns columns [w*512,(w+1)*512): 16 groups of 2 N-tiles; per K-step
// two A frags (LDS) x two B frags (global) feed 4 v_wmma_f32_16x16x32_f16.
//
// Fragment addressing (wave32):
//  A (16x32 f16, ISA table): lanes 0-15 row=lid hold K {kb..kb+7, kb+16..23},
//                            lanes 16-31 row=lid hold K {kb+8..15, kb+24..31}.
//  B (32x16 f16):            lanes 0-15 col=lid hold K {kb..kb+7, kb+16..23};
//                            lanes 16-31 col=lid hold K {kb+8..15, kb+24..31}.
//  C/D (16x16 f32): lanes 0-15 n=lid, vgpr j -> row j; lanes 16-31 n=lid,
//                            vgpr j -> row j+8.
// ---------------------------------------------------------------------------
__global__ __launch_bounds__(256) void infonce_fused(
    const _Float16* __restrict__ G, const _Float16* __restrict__ R,
    float* __restrict__ rowLoss, int D)
{
  constexpr int AS_STRIDE = 4104;              // 4096 + 8 halves of pad
  __shared__ _Float16 As[32 * AS_STRIDE];      // ~256.5 KB of the 320KB WGP LDS
  __shared__ float sMax[8][32];
  __shared__ float sSum[8][32];
  __shared__ float sDiag[32];

  const int t     = threadIdx.x;
  const int lane  = t & 31;
  const int wave  = t >> 5;                    // 0..7
  const int lid   = lane & 15;
  const int hilo  = lane >> 4;                 // 0: lanes 0-15, 1: lanes 16-31
  const int mBase = blockIdx.x * 32;           // global row base (32-aligned)

  // Cooperative copy of the 32-row A tile (contiguous 256KB) into LDS.
  {
    const uint4* src = (const uint4*)(G + (size_t)mBase * (size_t)D);
    for (int i = t; i < 32 * 512; i += 256) {  // 512 uint4 per row
      const int row = i >> 9, col = i & 511;
      *(uint4*)(As + row * AS_STRIDE + col * 8) = src[i];
    }
  }
  __syncthreads();

  // Per-lane running softmax stats; tile0 = rows 0-15, tile1 = rows 16-31.
  float rm0[8], rs0[8], rm1[8], rs1[8];
#pragma unroll
  for (int j = 0; j < 8; ++j) {
    rm0[j] = -3.0e38f; rs0[j] = 0.f;
    rm1[j] = -3.0e38f; rs1[j] = 0.f;
  }

  const _Float16* A0 = As + lid * AS_STRIDE + hilo * 8;          // rows 0-15
  const _Float16* A1 = As + (16 + lid) * AS_STRIDE + hilo * 8;   // rows 16-31

  for (int ng = 0; ng < 16; ++ng) {            // 16 groups of 2 N-tiles
    const int colBase = wave * 512 + ng * 32;
    const _Float16* B0 = R + (size_t)(colBase + lid) * (size_t)D + hilo * 8;
    const _Float16* B1 = B0 + (size_t)16 * (size_t)D;

    // Speculative prefetch of the next group's first B lines (L0 warm-up).
    if (ng < 15) __builtin_prefetch(B0 + (size_t)32 * (size_t)D, 0, 1);

    v8f c00 = {0.f,0.f,0.f,0.f,0.f,0.f,0.f,0.f};
    v8f c01 = c00, c10 = c00, c11 = c00;       // c[mTile][nTile]

#pragma unroll 2
    for (int kt = 0; kt < 128; ++kt) {
      const int kb = kt * 32;
      v16h a0 = cat16(*(const v8h*)(A0 + kb), *(const v8h*)(A0 + kb + 16));
      v16h a1 = cat16(*(const v8h*)(A1 + kb), *(const v8h*)(A1 + kb + 16));
      v16h b0 = cat16(*(const v8h*)(B0 + kb), *(const v8h*)(B0 + kb + 16));
      v16h b1 = cat16(*(const v8h*)(B1 + kb), *(const v8h*)(B1 + kb + 16));
      c00 = __builtin_amdgcn_wmma_f32_16x16x32_f16(false, a0, false, b0, (short)0, c00, false, false);
      c01 = __builtin_amdgcn_wmma_f32_16x16x32_f16(false, a0, false, b1, (short)0, c01, false, false);
      c10 = __builtin_amdgcn_wmma_f32_16x16x32_f16(false, a1, false, b0, (short)0, c10, false, false);
      c11 = __builtin_amdgcn_wmma_f32_16x16x32_f16(false, a1, false, b1, (short)0, c11, false, false);
    }

    // Online softmax updates (per-lane; merged across lanes at the end).
    auto upd = [&](v8f cc, float* m, float* s) {
#pragma unroll
      for (int j = 0; j < 8; ++j) {
        float v = cc[j] * TEMP_INV;
        float mn = fmaxf(m[j], v);
        s[j] = s[j] * __expf(m[j] - mn) + __expf(v - mn);
        m[j] = mn;
      }
    };
    upd(c00, rm0, rs0); upd(c01, rm0, rs0);
    upd(c10, rm1, rs1); upd(c11, rm1, rs1);

    // Diagonal: mBase and colBase are both 32-aligned, so the diagonal of
    // this block's rows sits exactly in (tile0,n0) and (tile1,n1) of the
    // group with colBase == mBase (uniform condition, one wave/group hits).
    if (colBase == mBase) {
#pragma unroll
      for (int j = 0; j < 8; ++j) {
        if (lid == hilo * 8 + j) {
          sDiag[hilo * 8 + j]      = c00[j] * TEMP_INV;  // rows 0-15
          sDiag[16 + hilo * 8 + j] = c11[j] * TEMP_INV;  // rows 16-31
        }
      }
    }
  }

  // Merge (m,s) across the 16 lanes of each half (xor<16 stays in-half).
  auto laneMerge = [&](float* m, float* s) {
#pragma unroll
    for (int off = 1; off < 16; off <<= 1) {
#pragma unroll
      for (int j = 0; j < 8; ++j) {
        float mo = __shfl_xor(m[j], off, 32);
        float so = __shfl_xor(s[j], off, 32);
        float mn = fmaxf(m[j], mo);
        s[j] = s[j] * __expf(m[j] - mn) + so * __expf(mo - mn);
        m[j] = mn;
      }
    }
  };
  laneMerge(rm0, rs0);
  laneMerge(rm1, rs1);

  if (lid == 0) {
#pragma unroll
    for (int j = 0; j < 8; ++j) {
      sMax[wave][hilo * 8 + j]      = rm0[j];
      sSum[wave][hilo * 8 + j]      = rs0[j];
      sMax[wave][16 + hilo * 8 + j] = rm1[j];
      sSum[wave][16 + hilo * 8 + j] = rs1[j];
    }
  }
  __syncthreads();

  // Combine the 8 wave-slabs per row; loss_i = lse_i - logit_ii.
  if (t < 32) {
    const int r = t;
    float M = -3.0e38f, S = 0.f;
#pragma unroll
    for (int w = 0; w < 8; ++w) {
      float mo = sMax[w][r], so = sSum[w][r];
      float mn = fmaxf(M, mo);
      S = S * __expf(M - mn) + so * __expf(mo - mn);
      M = mn;
    }
    rowLoss[mBase + r] = M + __logf(S) - sDiag[r];
  }
}

// ---------------------------------------------------------------------------
// Kernel 3: mean of per-row losses -> scalar.
// ---------------------------------------------------------------------------
__global__ __launch_bounds__(256) void reduce_mean(
    const float* __restrict__ rowLoss, float* __restrict__ out, int B)
{
  float s = 0.f;
  for (int i = threadIdx.x; i < B; i += 256) s += rowLoss[i];
#pragma unroll
  for (int off = 16; off; off >>= 1) s += __shfl_xor(s, off, 32);
  __shared__ float red[8];
  if ((threadIdx.x & 31) == 0) red[threadIdx.x >> 5] = s;
  __syncthreads();
  if (threadIdx.x == 0) {
    float t = 0.f;
#pragma unroll
    for (int w = 0; w < 8; ++w) t += red[w];
    out[0] = t / (float)B;
  }
}

// ---------------------------------------------------------------------------
extern "C" void kernel_launch(void* const* d_in, const int* in_sizes, int n_in,
                              void* d_out, int out_size, void* d_ws, size_t ws_size,
                              hipStream_t stream) {
  const float* gen  = (const float*)d_in[0];
  const float* real = (const float*)d_in[1];
  const int D = 4096;
  const int B = in_sizes[0] / D;   // 4096

  // Workspace layout: gH[B*D] f16 | rH[B*D] f16 | rowLoss[B] f32  (~64 MiB)
  _Float16* gH = (_Float16*)d_ws;
  _Float16* rH = gH + (size_t)B * (size_t)D;
  float* rowLoss = (float*)(rH + (size_t)B * (size_t)D);

  nrm_f32_to_f16<<<dim3(B, 2), 256, 0, stream>>>(gen, real, gH, rH, D);
  infonce_fused<<<dim3(B / 32), 256, 0, stream>>>(gH, rH, rowLoss, D);
  reduce_mean<<<1, 256, 0, stream>>>(rowLoss, (float*)d_out, B);
}
